// ALiBiGroupedQueryAttention_76493367541795
// MI455X (gfx1250) — matrix operand order repaired and checked by reading, not compiled, not run on hardware
//
#include <hip/hip_runtime.h>
#include <cstdint>
#include <cstddef>

// ---------------------------------------------------------------------------
// ALiBi GQA attention for gfx1250 (MI455X), bf16 WMMA path.
// B=4, S=1024, D_MODEL=2048, 16 Q heads, 4 KV heads, head_dim=128.
// ---------------------------------------------------------------------------

typedef __bf16 bf16;
typedef __attribute__((ext_vector_type(8)))  __bf16 v8bf;
typedef __attribute__((ext_vector_type(16))) __bf16 v16bf;
typedef __attribute__((ext_vector_type(8)))  float  v8f;
typedef __attribute__((ext_vector_type(4)))  int    v4i;

#define D_MODEL 2048
#define NQH     16
#define NKVH    4
#define HD      128
#define BB      4
#define SS      1024
#define KVD     512      // NKVH * HD
#define NEGINF  (-3.0e38f)

#if __has_builtin(__builtin_amdgcn_global_load_async_to_lds_b128)
#define GQA_ASYNC_LDS 1
#else
#define GQA_ASYNC_LDS 0
#endif

// Per-lane async global->LDS 16B transfer (GLOBAL_LOAD_ASYNC_TO_LDS_B128).
// Generic LDS pointer low 32 bits are the LDS byte address (ISA 10.2).
__device__ __forceinline__ void async_copy_b128(const void* g, void* lds) {
#if GQA_ASYNC_LDS
  typedef __attribute__((address_space(1))) v4i gv4i;   // global v4i
  typedef __attribute__((address_space(3))) v4i lv4i;   // LDS v4i
  __builtin_amdgcn_global_load_async_to_lds_b128(
      (gv4i*)(uintptr_t)g,
      (lv4i*)(unsigned)(uintptr_t)lds, 0, 0);
#else
  (void)g; (void)lds;
#endif
}

__device__ __forceinline__ void wait_async0() {
#if GQA_ASYNC_LDS
#if __has_builtin(__builtin_amdgcn_s_wait_asynccnt)
  __builtin_amdgcn_s_wait_asynccnt(0);
#else
  asm volatile("s_wait_asynccnt 0" ::: "memory");
#endif
#endif
}

__device__ __forceinline__ bf16 f2bf(float f) {
  unsigned u = __float_as_uint(f);
  unsigned r = (u + 0x7FFFu + ((u >> 16) & 1u)) >> 16;   // RNE
  unsigned short h = (unsigned short)r;
  return __builtin_bit_cast(bf16, h);
}

__device__ __forceinline__ v16bf cat8(v8bf lo, v8bf hi) {
  return __builtin_shufflevector(lo, hi, 0,1,2,3,4,5,6,7,8,9,10,11,12,13,14,15);
}

__device__ __forceinline__ v8f wmma_bf16(v16bf a, v16bf b, v8f c) {
  // D = A(16x32 bf16) * B(32x16 bf16) + C(16x16 f32)
  return __builtin_amdgcn_wmma_f32_16x16x32_bf16(false, a, false, b, (short)0, c,
                                                 false, false);
}

// ---------------------------------------------------------------------------
// fp32 -> bf16 conversion (grid-stride)
// ---------------------------------------------------------------------------
__global__ void gqa_cvt_f32_bf16(const float* __restrict__ in,
                                 bf16* __restrict__ out, size_t n) {
  size_t i = (size_t)blockIdx.x * blockDim.x + threadIdx.x;
  size_t stride = (size_t)gridDim.x * blockDim.x;
  for (; i < n; i += stride) out[i] = f2bf(in[i]);
}

// ---------------------------------------------------------------------------
// Tiled bf16 GEMM: C[M,N] = A[M,K] * B[K,N], f32 accumulate.
// Block = 256 threads (8 waves). Block tile 128x128, TILE_K = 64.
// Wave grid 2(M) x 4(N): each wave owns 64x32 = 4x2 WMMA tiles.
// A tile staged via async global->LDS (ASYNCcnt); B tile staged transposed
// through VGPRs (needs the transpose anyway). Next tiles prefetched.
// M, N, K assumed multiples of 128/128/64 (true for this problem).
// ---------------------------------------------------------------------------
__global__ __launch_bounds__(256) void gqa_gemm_bf16(
    const bf16* __restrict__ A, const bf16* __restrict__ B,
    void* __restrict__ C, int M, int N, int K, int out_f32)
{
  __shared__ bf16 As[128][72];   // [m][k], pad 64->72 (row = 144B, 16B aligned)
  __shared__ bf16 Bs[128][72];   // transposed: [n][k]

  const int tid  = threadIdx.x;
  const int lane = tid & 31;
  const int wave = tid >> 5;
  const int l16  = lane & 15;
  const int half = lane >> 4;        // 0 or 1
  const int wm   = wave >> 2;        // 0..1
  const int wn   = wave & 3;         // 0..3
  const int m0   = blockIdx.y * 128;
  const int n0   = blockIdx.x * 128;

  const v8f vzero = {0.f,0.f,0.f,0.f,0.f,0.f,0.f,0.f};
  v8f acc[4][2];
  #pragma unroll
  for (int i = 0; i < 4; ++i) { acc[i][0] = vzero; acc[i][1] = vzero; }

  const int ar  = tid >> 1,  ac0 = (tid & 1) * 32;   // A: 128 rows x 64 cols
  const int bk  = tid >> 2,  bn0 = (tid & 3) * 32;   // B: 64 rows x 128 cols

  for (int k0 = 0; k0 < K; k0 += 64) {
    // stage A tile (row-major) via async DMA to this thread's LDS slots
    const bf16* ap = A + (size_t)(m0 + ar) * K + k0 + ac0;
#if GQA_ASYNC_LDS
    #pragma unroll
    for (int u = 0; u < 4; ++u)
      async_copy_b128(ap + u*8, &As[ar][ac0 + u*8]);
#else
    #pragma unroll
    for (int u = 0; u < 4; ++u)
      *(v8bf*)&As[ar][ac0 + u*8] = *(const v8bf*)(ap + u*8);
#endif
    // stage B tile transposed: Bs[n][k] (overlaps with async A transfers)
    const bf16* bp = B + (size_t)(k0 + bk) * N + n0 + bn0;
    #pragma unroll
    for (int u = 0; u < 4; ++u) {
      v8bf t = *(const v8bf*)(bp + u*8);
      #pragma unroll
      for (int e = 0; e < 8; ++e) Bs[bn0 + u*8 + e][bk] = t[e];
    }
    // prefetch next K tile while this stage computes
    if (k0 + 64 < K) {
      __builtin_prefetch(ap + 64, 0, 1);
      __builtin_prefetch(bp + (size_t)64 * N, 0, 1);
    }
    wait_async0();
    __syncthreads();

    #pragma unroll
    for (int ks = 0; ks < 2; ++ks) {
      const int kc = ks * 32;
      v16bf af[4];
      #pragma unroll
      for (int mt = 0; mt < 4; ++mt) {
        const int m = wm*64 + mt*16 + l16;           // A row = lane&15
        v8bf lo = *(const v8bf*)&As[m][kc + half*8];       // K = kb..kb+7
        v8bf hi = *(const v8bf*)&As[m][kc + half*8 + 16];  // K = kb+16..kb+23
        af[mt] = cat8(lo, hi);
      }
      v16bf bfg[2];
      #pragma unroll
      for (int nt = 0; nt < 2; ++nt) {
        const int n = wn*32 + nt*16 + l16;           // B col = lane&15
        v8bf lo = *(const v8bf*)&Bs[n][kc + half*8];       // K = 0..7 / 8..15
        v8bf hi = *(const v8bf*)&Bs[n][kc + half*8 + 16];  // K = 16..23 / 24..31
        bfg[nt] = cat8(lo, hi);
      }
      #pragma unroll
      for (int mt = 0; mt < 4; ++mt)
        #pragma unroll
        for (int nt = 0; nt < 2; ++nt)
          acc[mt][nt] = wmma_bf16(af[mt], bfg[nt], acc[mt][nt]);
    }
    __syncthreads();
  }

  // C layout: lane holds column l16; rows = half*8 + i
  #pragma unroll
  for (int mt = 0; mt < 4; ++mt)
    #pragma unroll
    for (int nt = 0; nt < 2; ++nt)
      #pragma unroll
      for (int i = 0; i < 8; ++i) {
        const int m = m0 + wm*64 + mt*16 + half*8 + i;
        const int n = n0 + wn*32 + nt*16 + l16;
        const float v = acc[mt][nt][i];
        if (out_f32) ((float*)C)[(size_t)m * N + n] = v;
        else         ((bf16*)C)[(size_t)m * N + n]  = f2bf(v);
      }
}

// ---------------------------------------------------------------------------
// Flash-style GQA attention with ALiBi + padding mask.
// Grid: (S/64, NQH, B). Block: 128 threads = 4 waves; wave w owns q rows
// [qblk*64 + w*16, +16). KV processed in chunks of 32 with online softmax.
// ---------------------------------------------------------------------------
__global__ __launch_bounds__(128) void gqa_attn(
    const bf16* __restrict__ Qp,    // [B*S, 2048]  (head h at col h*128)
    const bf16* __restrict__ Kp,    // [B*S, 512]   (group g at col g*128)
    const bf16* __restrict__ Vp,    // [B*S, 512]
    const unsigned char* __restrict__ mask,  // [B, S] bool
    bf16* __restrict__ attn)        // [B*S, 2048]
{
  __shared__ bf16 Vt[HD][40];      // V chunk transposed: [d][kv], pad 32->40
  __shared__ bf16 Pb[4][16][40];   // per-wave P staging (16 x 32)

  const int lane = threadIdx.x & 31;
  const int wave = threadIdx.x >> 5;
  const int l16  = lane & 15;
  const int half = lane >> 4;
  const int b    = blockIdx.z;
  const int h    = blockIdx.y;
  const int qblk = blockIdx.x;
  const int g    = h >> 2;                       // GQA group
  const int qrow0 = qblk * 64 + wave * 16;
  const float slope = exp2f(-0.5f * (float)(h + 1));   // ALiBi: 2^(-8(h+1)/16)
  const float scale = 0.0883883476483184f;             // 1/sqrt(128)

  // Preload Q fragments (A layout) straight from global: 4 k-chunks of 32.
  v16bf qf[4];
  const bf16* qbase = Qp + ((size_t)(b * SS + qrow0 + l16)) * D_MODEL + h * HD;
  #pragma unroll
  for (int dc = 0; dc < 4; ++dc) {
    const int kb = dc * 32 + half * 8;
    v8bf lo = *(const v8bf*)(qbase + kb);
    v8bf hi = *(const v8bf*)(qbase + kb + 16);
    qf[dc] = cat8(lo, hi);
  }

  const v8f vzero = {0.f,0.f,0.f,0.f,0.f,0.f,0.f,0.f};
  v8f   o[8];                     // 16 x 128 output accumulator (8 n-subtiles)
  float rm[8], rl[8];             // per-row running max / sum (rows half*8+i)
  #pragma unroll
  for (int i = 0; i < 8; ++i) { o[i] = vzero; rm[i] = NEGINF; rl[i] = 0.f; }

  for (int kv0 = 0; kv0 < SS; kv0 += 32) {
    // ---- stage V chunk transposed into LDS: Vt[d][kv_local] ----
    {
      const int kvl = threadIdx.x >> 2;          // 0..31
      const int d0  = (threadIdx.x & 3) * 32;    // 0,32,64,96
      const bf16* vp = Vp + ((size_t)(b * SS + kv0 + kvl)) * KVD + g * HD + d0;
      #pragma unroll
      for (int u = 0; u < 4; ++u) {
        v8bf t = *(const v8bf*)(vp + u*8);
        #pragma unroll
        for (int e = 0; e < 8; ++e) Vt[d0 + u*8 + e][kvl] = t[e];
      }
      if (kv0 + 32 < SS) {
        __builtin_prefetch(vp + (size_t)32 * KVD, 0, 1);                 // next V
        __builtin_prefetch(Kp + ((size_t)(b * SS + kv0 + 32 + kvl)) * KVD
                              + g * HD + d0, 0, 1);                      // next K
      }
    }
    __syncthreads();

    // ---- scores: S(16x32) = Q(16x128) * K^T(128x32), K frags from global ----
    v8f sa[2] = { vzero, vzero };
    #pragma unroll
    for (int dc = 0; dc < 4; ++dc) {
      #pragma unroll
      for (int nt = 0; nt < 2; ++nt) {
        const int kvcol = kv0 + nt * 16 + l16;   // B col = kv index
        const bf16* kp = Kp + ((size_t)(b * SS + kvcol)) * KVD + g * HD
                            + dc * 32 + half * 8;
        v8bf lo = *(const v8bf*)kp;
        v8bf hi = *(const v8bf*)(kp + 16);
        sa[nt] = wmma_bf16(qf[dc], cat8(lo, hi), sa[nt]);
      }
    }

    // ---- bias + mask + online softmax (rows live in 16-lane halves) ----
    const int j0 = kv0 + l16, j1 = kv0 + 16 + l16;
    const bool mk0 = mask[(size_t)b * SS + j0] != 0;
    const bool mk1 = mask[(size_t)b * SS + j1] != 0;
    #pragma unroll
    for (int i = 0; i < 8; ++i) {
      const int iq = qrow0 + half * 8 + i;
      float s0 = sa[0][i] * scale + slope * (float)(j0 - iq);
      float s1 = sa[1][i] * scale + slope * (float)(j1 - iq);
      s0 = mk0 ? s0 : NEGINF;
      s1 = mk1 ? s1 : NEGINF;
      float mx = fmaxf(s0, s1);
      #pragma unroll
      for (int off = 8; off >= 1; off >>= 1)
        mx = fmaxf(mx, __shfl_xor(mx, off, 32));   // stays within the half
      const float nm   = fmaxf(rm[i], mx);
      const float corr = __expf(rm[i] - nm);
      const float e0 = __expf(s0 - nm);
      const float e1 = __expf(s1 - nm);
      float rs = e0 + e1;
      #pragma unroll
      for (int off = 8; off >= 1; off >>= 1)
        rs += __shfl_xor(rs, off, 32);
      rl[i] = rl[i] * corr + rs;
      rm[i] = nm;
      #pragma unroll
      for (int ns = 0; ns < 8; ++ns) o[ns][i] *= corr;
      // stage P in C layout; only this wave touches Pb[wave]
      Pb[wave][half * 8 + i][l16]      = f2bf(e0);
      Pb[wave][half * 8 + i][16 + l16] = f2bf(e1);
    }
    asm volatile("s_wait_dscnt 0" ::: "memory");   // order P store -> P load

    // ---- O += P(16x32) * V(32x128): P from LDS (A layout), V from Vt ----
    v16bf pf;
    {
      v8bf lo = *(const v8bf*)&Pb[wave][l16][half * 8];
      v8bf hi = *(const v8bf*)&Pb[wave][l16][half * 8 + 16];
      pf = cat8(lo, hi);
    }
    #pragma unroll
    for (int ns = 0; ns < 8; ++ns) {
      const int n = ns * 16 + l16;                 // B col = d index
      v8bf lo = *(const v8bf*)&Vt[n][half * 8];
      v8bf hi = *(const v8bf*)&Vt[n][half * 8 + 16];
      o[ns] = wmma_bf16(pf, cat8(lo, hi), o[ns]);
    }
    __syncthreads();   // protect Vt before next chunk's staging
  }

  // ---- normalize and write bf16 attention output ----
  #pragma unroll
  for (int i = 0; i < 8; ++i) {
    const float inv = 1.0f / rl[i];
    const int q = qrow0 + half * 8 + i;
    bf16* op = attn + ((size_t)(b * SS + q)) * D_MODEL + h * HD;
    #pragma unroll
    for (int ns = 0; ns < 8; ++ns)
      op[ns * 16 + l16] = f2bf(o[ns][i] * inv);
  }
}

// ---------------------------------------------------------------------------
// Host-side orchestration
// ---------------------------------------------------------------------------
extern "C" void kernel_launch(void* const* d_in, const int* in_sizes, int n_in,
                              void* d_out, int out_size, void* d_ws, size_t ws_size,
                              hipStream_t stream) {
  const float* query = (const float*)d_in[0];
  const float* key   = (const float*)d_in[1];
  const float* value = (const float*)d_in[2];
  const unsigned char* mask = (const unsigned char*)d_in[3];
  const float* Wq = (const float*)d_in[4];
  const float* Wk = (const float*)d_in[5];
  const float* Wv = (const float*)d_in[6];
  const float* Wo = (const float*)d_in[7];
  float* out = (float*)d_out;

  const size_t NX  = (size_t)BB * SS * D_MODEL;     // 8,388,608
  const size_t NW  = (size_t)D_MODEL * D_MODEL;     // 4,194,304
  const size_t NWK = (size_t)D_MODEL * KVD;         // 1,048,576
  const size_t NKV = (size_t)BB * SS * KVD;         // 2,097,152

  size_t off = 0;
  auto carve = [&](size_t elems) -> bf16* {
    bf16* p = (bf16*)((char*)d_ws + off);
    off = (off + elems * sizeof(bf16) + 255) & ~(size_t)255;
    return p;
  };
  bf16* qb  = carve(NX);   bf16* kb  = carve(NX);   bf16* vb  = carve(NX);
  bf16* wqb = carve(NW);   bf16* wkb = carve(NWK);  bf16* wvb = carve(NWK);
  bf16* wob = carve(NW);
  bf16* Qp  = carve(NX);   bf16* Kp  = carve(NKV);  bf16* Vp  = carve(NKV);
  bf16* at  = carve(NX);
  (void)ws_size; // ~115 MB required

  // fp32 -> bf16
  const int CB = 256, CG = 2048;
  gqa_cvt_f32_bf16<<<CG, CB, 0, stream>>>(query, qb,  NX);
  gqa_cvt_f32_bf16<<<CG, CB, 0, stream>>>(key,   kb,  NX);
  gqa_cvt_f32_bf16<<<CG, CB, 0, stream>>>(value, vb,  NX);
  gqa_cvt_f32_bf16<<<CG, CB, 0, stream>>>(Wq,    wqb, NW);
  gqa_cvt_f32_bf16<<<CG, CB, 0, stream>>>(Wk,    wkb, NWK);
  gqa_cvt_f32_bf16<<<CG, CB, 0, stream>>>(Wv,    wvb, NWK);
  gqa_cvt_f32_bf16<<<CG, CB, 0, stream>>>(Wo,    wob, NW);

  const int M = BB * SS;  // 4096
  // Projections: Qp = q @ Wq ; Kp = k @ Wk ; Vp = v @ Wv  (bf16 out)
  gqa_gemm_bf16<<<dim3(D_MODEL/128, M/128), 256, 0, stream>>>(qb, wqb, Qp, M, D_MODEL, D_MODEL, 0);
  gqa_gemm_bf16<<<dim3(KVD/128,     M/128), 256, 0, stream>>>(kb, wkb, Kp, M, KVD,     D_MODEL, 0);
  gqa_gemm_bf16<<<dim3(KVD/128,     M/128), 256, 0, stream>>>(vb, wvb, Vp, M, KVD,     D_MODEL, 0);

  // Attention
  gqa_attn<<<dim3(SS/64, NQH, BB), 128, 0, stream>>>(Qp, Kp, Vp, mask, at);

  // Output projection (f32 out -> d_out)
  gqa_gemm_bf16<<<dim3(D_MODEL/128, M/128), 256, 0, stream>>>(at, wob, out, M, D_MODEL, D_MODEL, 1);

  (void)in_sizes; (void)n_in; (void)out_size;
}
